// BaseTextureNCA_20899310862985
// MI455X (gfx1250) — matrix operand order repaired and checked by reading, not compiled, not run on hardware
//
#include <hip/hip_runtime.h>

typedef __attribute__((ext_vector_type(16))) _Float16     v16h;
typedef __attribute__((ext_vector_type(8)))  float        v8f;
typedef __attribute__((ext_vector_type(8)))  unsigned int v8u;
typedef _Float16 h16;

#define HW    512
#define CHN   12
#define HID   96
#define NTHR  128   // 4 waves per workgroup
#define TPIX  256   // pixels per workgroup (16 M-tiles; 4 per wave -> weight frags amortized 2x)
#define PSTR  66    // perception row stride (halves): 132B = 33 dwords (odd -> conflict-free)
#define MSTR  24    // sHT pixel-dim stride (halves): 48B (16B-aligned rows for b128 stores)
#define DSTR  20    // sDyT pixel-dim stride (floats): 80B (16B-aligned rows)

union FragU { v16h h; v8u u; };

// 16-bit A-matrix 16x32 fragment (contiguous k order used here; B fragments use the same), source row is k-contiguous in LDS:
// lane<16: elems 0..15 -> K=k0..k0+15
// lane>=16: elems 0..15 -> K=k0+16..k0+31
__device__ __forceinline__ v16h ld_afrag(const h16* __restrict__ row, int k0, int hi) {
  FragU f;
  const unsigned int* p0 = (const unsigned int*)(row + k0 + hi * 16);
  const unsigned int* p1 = (const unsigned int*)(row + k0 + hi * 16 + 8);
#pragma unroll
  for (int i = 0; i < 4; ++i) { f.u[i] = p0[i]; f.u[4 + i] = p1[i]; }
  return f.h;
}

// Same A fragment but gathered from a transposed [K][M] LDS buffer (strided u16 reads).
__device__ __forceinline__ v16h ld_afragT(const h16* __restrict__ base, int m, int k0, int hi) {
  v16h f;
#pragma unroll
  for (int i = 0; i < 8; ++i) {
    f[i]     = base[(k0 + hi * 16 + i) * MSTR + m];
    f[8 + i] = base[(k0 + hi * 16 + 8 + i) * MSTR + m];
  }
  return f;
}

// 16-bit B-matrix 32x16 fragment built straight from global f32 weights (k-contiguous):
// lane = N within each 16-lane half; lanes 0-15 hold K=k0..k0+15, lanes 16-31 K=k0+16..31.
// `valid=false` lanes produce a zero fragment (padding rows/columns) with no OOB reads.
__device__ __forceinline__ v16h bfrag_global(const float* __restrict__ w, int off, bool valid) {
  v16h f;
#pragma unroll
  for (int e = 0; e < 16; ++e) {
    float v = valid ? w[off + e] : 0.f;
    f[e] = (h16)v;
  }
  return f;
}

__global__ __launch_bounds__(NTHR) void nca_fused_wmma(
    const float* __restrict__ x,    // [8,12,512,512]
    const float* __restrict__ w1w,  // [96,48]
    const float* __restrict__ w1b,  // [96]
    const float* __restrict__ w2w,  // [12,96]
    const float* __restrict__ ru,   // [8,1,512,512]
    float* __restrict__ out)        // [8,12,512,512]
{
  __shared__ __align__(16) h16   sPerc[TPIX][PSTR];     // perception [pix][feat 0..63]
  __shared__ __align__(16) h16   sHT[4][HID][MSTR];     // per-wave hidden, transposed [n][m]
  __shared__ __align__(16) float sDyT[4][16][DSTR];     // per-wave dy, transposed [ch][pix]
  __shared__ __align__(16) float sRow[3][260];          // 3 halo rows of one channel (258 used)

  const int tid = threadIdx.x;
  const int wg  = blockIdx.x;
  const int xt  = wg & 1;
  const int y   = (wg >> 1) & (HW - 1);
  const int b   = wg >> 10;
  const int x0  = xt * TPIX;

#pragma unroll
  for (int p = 0; p < 2; ++p)
#pragma unroll
    for (int k = 48; k < 64; ++k) sPerc[tid + p * NTHR][k] = (h16)0.f;  // K padding

  // ---- phase 1: perception (depthwise 3x3, circular pad), channel at a time ----
  for (int c = 0; c < CHN; ++c) {
    __syncthreads();  // protect sRow reuse
    const float* xc = x + (b * CHN + c) * HW * HW;
    for (int i = tid; i < 3 * 258; i += NTHR) {
      int r = i / 258, j = i % 258;
      int gr = (y + r - 1) & (HW - 1);
      int gc = (x0 + j - 1) & (HW - 1);
      sRow[r][j] = xc[gr * HW + gc];
      if (c + 1 < CHN) __builtin_prefetch(xc + HW * HW + gr * HW + gc, 0, 3);
    }
    __syncthreads();
#pragma unroll
    for (int p = 0; p < 2; ++p) {
      const int t = tid + p * NTHR;
      const int j = t + 1;
      float c00 = sRow[0][j - 1], c01 = sRow[0][j], c02 = sRow[0][j + 1];
      float c10 = sRow[1][j - 1], c11 = sRow[1][j], c12 = sRow[1][j + 1];
      float c20 = sRow[2][j - 1], c21 = sRow[2][j], c22 = sRow[2][j + 1];
      sPerc[t][4 * c + 0] = (h16)c11;                                             // identity
      sPerc[t][4 * c + 1] = (h16)((c02 - c00) + 2.f * (c12 - c10) + (c22 - c20)); // sobel_x
      sPerc[t][4 * c + 2] = (h16)((c20 - c00) + 2.f * (c21 - c01) + (c22 - c02)); // sobel_y
      sPerc[t][4 * c + 3] = (h16)(c00 + 2.f * c01 + c02 + 2.f * c10 - 12.f * c11 +
                                  2.f * c12 + c20 + 2.f * c21 + c22);             // laplacian
    }
  }
  __syncthreads();

  // ---- phase 2: WMMA GEMM1 (48->96, relu) then GEMM2 (96->12), per-wave M-tiles ----
  const int wave = tid >> 5;
  const int lane = tid & 31;
  const int l16  = lane & 15;
  const int hi   = lane >> 4;

  // Hoisted weight B-fragments in registers, built once per wave from global (L2-hot),
  // amortized over 4 M-tiles (64 pixels) per wave.
  v16h B1[6][2], B2[3];
#pragma unroll
  for (int nt = 0; nt < 6; ++nt) {
    const int n = nt * 16 + l16;
#pragma unroll
    for (int kc = 0; kc < 2; ++kc) {
      const int kb = kc * 32 + hi * 16;
      B1[nt][kc] = bfrag_global(w1w, n * 48 + kb, kb < 48);   // K padded 48 -> 64
    }
  }
#pragma unroll
  for (int kc = 0; kc < 3; ++kc) {
    const int kb = kc * 32 + hi * 16;
    B2[kc] = bfrag_global(w2w, l16 * HID + kb, l16 < CHN);    // N padded 12 -> 16
  }

  float bias[6];
#pragma unroll
  for (int nt = 0; nt < 6; ++nt) bias[nt] = w1b[nt * 16 + l16];

#pragma unroll
  for (int mi = 0; mi < 4; ++mi) {
    const int mt = wave + mi * 4;                 // M-tile index (0..15)
    const h16* prow = &sPerc[mt * 16 + l16][0];
    v16h a0 = ld_afrag(prow, 0, hi);
    v16h a1 = ld_afrag(prow, 32, hi);

#pragma unroll
    for (int nt = 0; nt < 6; ++nt) {
      v8f c = {};
      c = __builtin_amdgcn_wmma_f32_16x16x32_f16(false, a0, false, B1[nt][0], (short)0, c, false, false);
      c = __builtin_amdgcn_wmma_f32_16x16x32_f16(false, a1, false, B1[nt][1], (short)0, c, false, false);
      const float bv = bias[nt];
      // C layout: lane's 8 elems are M = hi*8 + r (contiguous), N = nt*16 + l16 fixed
      // -> pack ReLU(h) into one 16B store into the transposed [n][m] buffer.
      union { h16 h[8]; uint4 q; } pk;
#pragma unroll
      for (int r = 0; r < 8; ++r) {
        float hv = c[r] + bv;
        pk.h[r] = (h16)(hv > 0.f ? hv : 0.f);
      }
      *(uint4*)&sHT[wave][nt * 16 + l16][hi * 8] = pk.q;
    }

    // GEMM2: K = 96 = 3 x 32 (same-wave LDS is in-order; no barrier needed)
    v8f d = {};
    const h16* hbase = &sHT[wave][0][0];
#pragma unroll
    for (int kc = 0; kc < 3; ++kc) {
      v16h ah = ld_afragT(hbase, l16, kc * 32, hi);
      d = __builtin_amdgcn_wmma_f32_16x16x32_f16(false, ah, false, B2[kc], (short)0, d, false, false);
    }
    // D layout: lane = channel (l16), elems = pixels hi*8+r -> two packed 16B stores.
    float4 q0 = make_float4(d[0], d[1], d[2], d[3]);
    float4 q1 = make_float4(d[4], d[5], d[6], d[7]);
    *(float4*)&sDyT[wave][l16][hi * 8]     = q0;
    *(float4*)&sDyT[wave][l16][hi * 8 + 4] = q1;

    // epilogue: coalesced x/rand loads + out stores, stochastic mask (one mask load per pixel)
    const int pbase = x0 + mt * 16;
    const float mval = floorf(ru[(b * HW + y) * HW + pbase + l16] + 0.5f);
#pragma unroll
    for (int it = 0; it < 6; ++it) {
      const int ch  = it * 2 + hi;                // 0..11
      const int gidx = ((b * CHN + ch) * HW + y) * HW + pbase + l16;
      out[gidx] = x[gidx] + sDyT[wave][ch][l16] * mval;
    }
  }
}

extern "C" void kernel_launch(void* const* d_in, const int* in_sizes, int n_in,
                              void* d_out, int out_size, void* d_ws, size_t ws_size,
                              hipStream_t stream) {
  (void)in_sizes; (void)n_in; (void)d_ws; (void)ws_size; (void)out_size;
  const float* x    = (const float*)d_in[0];
  const float* w1w  = (const float*)d_in[1];
  const float* w1b  = (const float*)d_in[2];
  const float* w2w  = (const float*)d_in[3];
  const float* ru   = (const float*)d_in[4];
  float* out        = (float*)d_out;

  const int blocks = 8 * HW * (HW / TPIX);  // 8192 workgroups, one 256-pixel row segment each
  nca_fused_wmma<<<blocks, NTHR, 0, stream>>>(x, w1w, w1b, w2w, ru, out);
}
